// InterestExtractLayer_11141145166029
// MI455X (gfx1250) — matrix operand order repaired and checked by reading, not compiled, not run on hardware
//
#include <hip/hip_runtime.h>
#include <hip/hip_bf16.h>

typedef _Float16 f16;
typedef __attribute__((ext_vector_type(16))) _Float16 v16h;
typedef __attribute__((ext_vector_type(8)))  _Float16 h8;
typedef __attribute__((ext_vector_type(8)))  float    v8f;
typedef __attribute__((ext_vector_type(4)))  float    f4;

// Problem dims (fixed by setup_inputs)
constexpr int Bb = 1024, Tt = 200, Ee = 128, N3 = 384;
constexpr int Tm = Tt - 1;              // 199
constexpr int NR = Bb * Tm;             // 203776 sequence rows for aux MLP
constexpr long OUT_GRU = (long)Bb * Tt * Ee;

constexpr float LOG_E   = -16.118095650958320f;   // log(1e-7)
constexpr float LOG_1ME = -1.0000000494736474e-7f; // log(1 - 1e-7)

// ---------------- WMMA helpers (CDNA5 16x16x32 f16 -> f32) ----------------
__device__ __forceinline__ v8f wmma16(v16h a, v16h b, v8f c) {
  return __builtin_amdgcn_wmma_f32_16x16x32_f16(false, a, false, b,
                                                (short)0, c, false, false);
}
// A fragment: lane<16 holds row=lane, K {k0..k0+7, k0+16..k0+23}; lanes>=16 shift k0 by 8.
__device__ __forceinline__ v16h mk_a(const f16* p) {
  h8 lo = *(const h8*)p;
  h8 hi = *(const h8*)(p + 16);
  return __builtin_shufflevector(lo, hi, 0,1,2,3,4,5,6,7,8,9,10,11,12,13,14,15);
}
__device__ __forceinline__ h8 cvt8(const float* p) {
  f4 a = *(const f4*)p;
  f4 b = *(const f4*)(p + 4);
  h8 r;
  r[0]=(f16)a[0]; r[1]=(f16)a[1]; r[2]=(f16)a[2]; r[3]=(f16)a[3];
  r[4]=(f16)b[0]; r[5]=(f16)b[1]; r[6]=(f16)b[2]; r[7]=(f16)b[3];
  return r;
}
__device__ __forceinline__ v16h mk_a_f32(const float* p) {
  h8 lo = cvt8(p);
  h8 hi = cvt8(p + 16);
  return __builtin_shufflevector(lo, hi, 0,1,2,3,4,5,6,7,8,9,10,11,12,13,14,15);
}
__device__ __forceinline__ float sigmoidf_(float t) { return 1.f / (1.f + __expf(-t)); }

// ---------------- GRU scan kernel ----------------
// One WG = 16 batch rows, full T=200 scan. Weights resident in LDS as f16.
constexpr int GRU_OFF_WK = 0;        // f16 [384][128]  (N-major)
constexpr int GRU_OFF_WR = 98304;    // f16 [384][128]
constexpr int GRU_OFF_XB = 196608;   // f16 [16][128]
constexpr int GRU_OFF_HB = 200704;   // f16 [16][128]
constexpr int GRU_OFF_HF = 204800;   // f32 [16][128]
constexpr int GRU_OFF_SX = 212992;   // f32 [16][384]
constexpr int GRU_OFF_SR = 237568;   // f32 [16][384]
constexpr int GRU_OFF_BI = 262144;   // f32 [384]
constexpr int GRU_OFF_BR = 263680;   // f32 [384]
constexpr int GRU_LDS    = 265216;   // <= 320KB/WGP per CDNA5 ISA

__global__ void gru_scan_kernel(const float* __restrict__ x,
                                const int*   __restrict__ mask,
                                const float* __restrict__ gk,
                                const float* __restrict__ grk,
                                const float* __restrict__ gb,
                                float* __restrict__ out) {
  extern __shared__ char smem[];
  f16*   wK = (f16*)(smem + GRU_OFF_WK);
  f16*   wR = (f16*)(smem + GRU_OFF_WR);
  f16*   xB = (f16*)(smem + GRU_OFF_XB);
  f16*   hB = (f16*)(smem + GRU_OFF_HB);
  float* hF = (float*)(smem + GRU_OFF_HF);
  float* sx = (float*)(smem + GRU_OFF_SX);
  float* sr = (float*)(smem + GRU_OFF_SR);
  float* bI = (float*)(smem + GRU_OFF_BI);
  float* bR = (float*)(smem + GRU_OFF_BR);

  const int tid   = threadIdx.x;
  const int bbase = blockIdx.x * 16;

  // Load + transpose weights to N-major f16 (one-time, L2-resident source).
  for (int idx = tid; idx < Ee * N3; idx += 256) {
    int k = idx / N3, n = idx % N3;
    wK[n * Ee + k] = (f16)gk[idx];
    wR[n * Ee + k] = (f16)grk[idx];
  }
  for (int n = tid; n < N3; n += 256) { bI[n] = gb[n]; bR[n] = gb[N3 + n]; }
  for (int i = tid; i < 16 * Ee; i += 256) { hB[i] = (f16)0.f; hF[i] = 0.f; }
  __syncthreads();

  const int lane = tid & 31;
  const int wave = tid >> 5;
  const int sel  = lane >> 4;   // K-half selector
  const int rc   = lane & 15;   // row (A/D) or col (B)
  const int lrow = tid >> 4;    // loader/elementwise row
  const int lcol = (tid & 15) * 8;

  for (int t = 0; t < Tt; ++t) {
    // Stage x_t tile -> f16 LDS
    {
      const float* src = x + ((size_t)(bbase + lrow) * Tt + t) * Ee + lcol;
      f4 a = *(const f4*)src;
      f4 b = *(const f4*)(src + 4);
      f16* dst = xB + lrow * Ee + lcol;
      dst[0]=(f16)a[0]; dst[1]=(f16)a[1]; dst[2]=(f16)a[2]; dst[3]=(f16)a[3];
      dst[4]=(f16)b[0]; dst[5]=(f16)b[1]; dst[6]=(f16)b[2]; dst[7]=(f16)b[3];
      if (t + 1 < Tt) __builtin_prefetch(src + Ee, 0, 1);
    }
    __syncthreads();

    // xw_t = x_t @ K  and  rec = h @ R   (fused, WMMA)
    #pragma unroll
    for (int q = 0; q < 3; ++q) {
      const int nt = wave * 3 + q;
      v8f accX = {};
      v8f accH = {};
      #pragma unroll
      for (int kk = 0; kk < 4; ++kk) {
        const int ka = kk * 32 + sel * 8;
        const int kb = kk * 32 + sel * 16;
        v16h ax = mk_a(xB + rc * Ee + ka);
        v16h ah = mk_a(hB + rc * Ee + ka);
        v16h bk = mk_a(wK + (nt * 16 + rc) * Ee + ka);
        v16h br = mk_a(wR + (nt * 16 + rc) * Ee + ka);
        accX = wmma16(ax, bk, accX);
        accH = wmma16(ah, br, accH);
      }
      #pragma unroll
      for (int i = 0; i < 8; ++i) {
        const int m = i + sel * 8, n = nt * 16 + rc;
        sx[m * N3 + n] = accX[i];
        sr[m * N3 + n] = accH[i];
      }
    }
    __syncthreads();

    // Gate nonlinearities + masked state update (fp32 state in LDS)
    {
      const int  m0   = mask[(size_t)(bbase + lrow) * Tt + t];
      float* orow = out + ((size_t)(bbase + lrow) * Tt + t) * Ee;
      #pragma unroll
      for (int j = 0; j < 8; ++j) {
        const int e = lcol + j;
        float xz = sx[lrow * N3 + e]        + bI[e];
        float xr = sx[lrow * N3 + 128 + e]  + bI[128 + e];
        float xh = sx[lrow * N3 + 256 + e]  + bI[256 + e];
        float rz = sr[lrow * N3 + e]        + bR[e];
        float rr = sr[lrow * N3 + 128 + e]  + bR[128 + e];
        float rh = sr[lrow * N3 + 256 + e]  + bR[256 + e];
        float z  = sigmoidf_(xz + rz);
        float r  = sigmoidf_(xr + rr);
        float hh = tanhf(xh + r * rh);
        float hp = hF[lrow * Ee + e];
        float hn = z * hp + (1.f - z) * hh;
        float ho = (m0 > 0) ? hn : hp;
        hF[lrow * Ee + e] = ho;
        hB[lrow * Ee + e] = (f16)ho;
        orow[e] = ho;
      }
    }
    __syncthreads();
  }
}

// ---------------- Aux-loss MLP kernel (all 3 layers via WMMA) ----------------
constexpr int MLP_OFF_W1 = 0;        // f16 [112][256] (N-major, n>=100 zero)
constexpr int MLP_OFF_W2 = 57344;    // f16 [64][128]  (n>=50 or k>=100 zero)
constexpr int MLP_OFF_W3 = 73728;    // f16 [64]       (j>=50 zero)
constexpr int MLP_OFF_B1 = 73856;    // f32 [112]
constexpr int MLP_OFF_B2 = 74304;    // f32 [64]
constexpr int MLP_OFF_H1 = 74560;    // f16 per-wave [16][128] x 8 waves
constexpr int MLP_OFF_H2 = 107328;   // f16 per-wave [16][64]  x 8 waves
constexpr int MLP_LDS    = 123712;

__global__ void aux_mlp_kernel(const float* __restrict__ gru,
                               const float* __restrict__ x,
                               const float* __restrict__ negx,
                               const int*   __restrict__ nmask,
                               const float* __restrict__ w1, const float* __restrict__ b1,
                               const float* __restrict__ w2, const float* __restrict__ b2,
                               const float* __restrict__ w3, const float* __restrict__ b3,
                               float* __restrict__ accum) {
  extern __shared__ char smem[];
  f16*   W1 = (f16*)(smem + MLP_OFF_W1);
  f16*   W2 = (f16*)(smem + MLP_OFF_W2);
  f16*   W3 = (f16*)(smem + MLP_OFF_W3);
  float* B1 = (float*)(smem + MLP_OFF_B1);
  float* B2 = (float*)(smem + MLP_OFF_B2);

  const int tid  = threadIdx.x;
  const int lane = tid & 31;
  const int wave = tid >> 5;
  const int sel  = lane >> 4;
  const int rc   = lane & 15;

  // Zero-padded weights into LDS (padding makes extra N/K lanes inert).
  for (int idx = tid; idx < 112 * 256; idx += 256) {
    int n = idx >> 8, k = idx & 255;
    W1[idx] = (n < 100) ? (f16)w1[k * 100 + n] : (f16)0.f;
  }
  for (int idx = tid; idx < 64 * 128; idx += 256) {
    int n = idx >> 7, k = idx & 127;
    W2[idx] = (n < 50 && k < 100) ? (f16)w2[k * 50 + n] : (f16)0.f;
  }
  if (tid < 64)  W3[tid] = (tid < 50)  ? (f16)w3[tid] : (f16)0.f;
  if (tid < 112) B1[tid] = (tid < 100) ? b1[tid] : 0.f;
  if (tid < 64)  B2[tid] = (tid < 50)  ? b2[tid] : 0.f;
  __syncthreads();

  f16* H1 = (f16*)(smem + MLP_OFF_H1) + wave * 16 * 128;
  f16* H2 = (f16*)(smem + MLP_OFF_H2) + wave * 16 * 64;
  if (lane < 16)  // zero K-pad cols 112..127 of H1 once (never rewritten)
    for (int r = 0; r < 16; ++r) H1[r * 128 + 112 + lane] = (f16)0.f;

  const int gw  = blockIdx.x * 8 + wave;
  const int gws = gridDim.x * 8;
  const int ntiles = (NR + 15) / 16;
  const float b3v = b3[0];
  float lloss = 0.f, lnm = 0.f;

  for (int tile = gw; tile < ntiles; tile += gws) {
    const int s  = tile * 16 + rc;
    const int sc = (s < NR) ? s : 0;
    const int bb = sc / Tm, jj = sc % Tm;
    const float* g_row = gru  + ((size_t)bb * Tt + jj + 1) * Ee;  // gru_embed row
    const float* x_row = x    + ((size_t)bb * Tt + jj + 1) * Ee;  // pos second half
    const float* n_row = negx + ((size_t)bb * Tm + jj) * Ee;      // neg second half

    for (int br = 0; br < 2; ++br) {
      const float* half2 = br ? n_row : x_row;

      // Hoist the 8 A-fragments of the 16x256 input tile (global->f16 regs)
      v16h afr[8];
      #pragma unroll
      for (int kk = 0; kk < 8; ++kk) {
        const int k0 = kk * 32 + sel * 8;
        const float* base = (k0 < 128) ? (g_row + k0) : (half2 + (k0 - 128));
        afr[kk] = mk_a_f32(base);
      }

      // L1: 16x256 @ 256x112 (+b1, relu) -> H1
      #pragma unroll
      for (int nt = 0; nt < 7; ++nt) {
        v8f acc = {};
        #pragma unroll
        for (int kk = 0; kk < 8; ++kk) {
          v16h bf = mk_a(W1 + (nt * 16 + rc) * 256 + kk * 32 + sel * 8);
          acc = wmma16(afr[kk], bf, acc);
        }
        #pragma unroll
        for (int i = 0; i < 8; ++i) {
          const int m = i + sel * 8, n = nt * 16 + rc;
          float v = acc[i] + B1[n];
          H1[m * 128 + n] = (f16)(v > 0.f ? v : 0.f);
        }
      }

      // L2: 16x128 @ 128x64 (+b2, relu) -> H2
      #pragma unroll
      for (int nt = 0; nt < 4; ++nt) {
        v8f acc = {};
        #pragma unroll
        for (int kk = 0; kk < 4; ++kk) {
          v16h af = mk_a(H1 + rc * 128 + kk * 32 + sel * 8);
          v16h bf = mk_a(W2 + (nt * 16 + rc) * 128 + kk * 32 + sel * 8);
          acc = wmma16(af, bf, acc);
        }
        #pragma unroll
        for (int i = 0; i < 8; ++i) {
          const int m = i + sel * 8, n = nt * 16 + rc;
          float v = acc[i] + B2[n];
          H2[m * 64 + n] = (f16)(v > 0.f ? v : 0.f);
        }
      }

      // L3: 16x64 @ 64x16 where only B column 0 is nonzero -> logits in lanes 0/16
      v8f acc3 = {};
      #pragma unroll
      for (int kk = 0; kk < 2; ++kk) {
        v16h af = mk_a(H2 + rc * 64 + kk * 32 + sel * 8);
        v16h bf = {};
        if (rc == 0) bf = mk_a(W3 + kk * 32 + sel * 8);
        acc3 = wmma16(af, bf, acc3);
      }

      if (rc == 0) {
        #pragma unroll
        for (int i = 0; i < 8; ++i) {
          const int rrow = i + sel * 8;
          const int ss = tile * 16 + rrow;
          if (ss < NR) {
            const int b2i = ss / Tm, j2 = ss % Tm;
            const float nm = (nmask[(size_t)b2i * Tm + j2] > 0) ? 1.f : 0.f;
            const float p  = sigmoidf_(acc3[i] + b3v);
            const float loss = (br == 0)
                ? -(p * LOG_1ME + (1.f - p) * LOG_E)
                : -(p * LOG_E   + (1.f - p) * LOG_1ME);
            lloss += loss * nm;
            if (br == 0) lnm += nm;
          }
        }
      }
    }
  }
  if (rc == 0) {
    atomicAdd(&accum[0], lloss);
    atomicAdd(&accum[1], lnm);
  }
}

// ---------------- tiny init / finalize ----------------
__global__ void init_accum_kernel(float* acc) {
  if (threadIdx.x < 2) acc[threadIdx.x] = 0.f;
}
__global__ void finalize_kernel(const float* __restrict__ acc, float* __restrict__ out_scalar) {
  if (threadIdx.x == 0) out_scalar[0] = acc[0] / (2.f * acc[1]);
}

extern "C" void kernel_launch(void* const* d_in, const int* in_sizes, int n_in,
                              void* d_out, int out_size, void* d_ws, size_t ws_size,
                              hipStream_t stream) {
  (void)in_sizes; (void)n_in; (void)out_size; (void)ws_size;
  const float* x     = (const float*)d_in[0];
  const int*   mask  = (const int*)d_in[1];
  const float* negx  = (const float*)d_in[2];
  const int*   nmask = (const int*)d_in[3];
  const float* gk    = (const float*)d_in[4];
  const float* grk   = (const float*)d_in[5];
  const float* gb    = (const float*)d_in[6];
  const float* w1    = (const float*)d_in[7];
  const float* b1    = (const float*)d_in[8];
  const float* w2    = (const float*)d_in[9];
  const float* b2    = (const float*)d_in[10];
  const float* w3    = (const float*)d_in[11];
  const float* b3    = (const float*)d_in[12];
  float* out  = (float*)d_out;
  float* accm = (float*)d_ws;

  init_accum_kernel<<<1, 32, 0, stream>>>(accm);
  gru_scan_kernel<<<Bb / 16, 256, GRU_LDS, stream>>>(x, mask, gk, grk, gb, out);
  aux_mlp_kernel<<<256, 256, MLP_LDS, stream>>>(out, x, negx, nmask,
                                                w1, b1, w2, b2, w3, b3, accm);
  finalize_kernel<<<1, 32, 0, stream>>>(accm, out + OUT_GRU);
}
